// FeatureExtractorLayer_41566693491081
// MI455X (gfx1250) — matrix operand order repaired and checked by reading, not compiled, not run on hardware
//
#include <hip/hip_runtime.h>

// ---------------------------------------------------------------------------
// MI455X (gfx1250, wave32) implementation of FeatureExtractorLayer:
//   residual VQ (Q=4, K=1024, D=512) + stride-2 conv1d (k=5) + exact GELU.
// Both GEMMs (VQ distance matmul: 137 GFLOP; conv-as-GEMM: 43 GFLOP) run on
// v_wmma_f32_16x16x32_bf16. Codebook (1MB bf16 per stage) stays L2-resident.
// ---------------------------------------------------------------------------

#define BB   8
#define TT   4096
#define DD   512
#define QQ   4
#define KK_  1024
#define TOUT 2048

typedef __attribute__((ext_vector_type(16))) __bf16 v16bf;
typedef __attribute__((ext_vector_type(8)))  __bf16 v8bf;
typedef __attribute__((ext_vector_type(8)))  float  v8f;

union BFrag { v16bf v; v8bf h[2]; };

__device__ __forceinline__ unsigned short f2bf(float f) {
  unsigned int u = __float_as_uint(f);
  u += 0x7FFFu + ((u >> 16) & 1u);              // round-to-nearest-even
  return (unsigned short)(u >> 16);
}

// ---------------- d_out layout (floats): y | quantized_out | indices | loss
#define Y_OFF    0
#define QOUT_OFF (BB * TOUT * DD)                       // 8388608
#define IDX_OFF  (QOUT_OFF + BB * TT * DD)              // 25165824
#define LOSS_OFF (IDX_OFF + QQ * BB * TT)               // 25296896

// ---------------- ws layout (bytes)
#define RES_OFF    0u                                   // residual f32   64 MB
#define CB16_OFF   67108864u                            // codebooks bf16  4 MB
#define CNORM_OFF  71303168u                            // ||c||^2 f32    16 KB
#define WT_OFF     71319552u                            // conv_w bf16   2.5 MB
#define Q16_OFF    73940992u                            // qout bf16      32 MB
// total ~107.5 MB

// ===========================================================================
__global__ void zero_loss_kernel(float* loss) { *loss = 0.0f; }

// one block per codebook row: convert f32 -> bf16 and compute ||c||^2
__global__ __launch_bounds__(128)
void prep_codebook_kernel(const float* __restrict__ cb,
                          unsigned short* __restrict__ cb16,
                          float* __restrict__ cnorm) {
  __shared__ float red[4];
  const int c = blockIdx.x, t = threadIdx.x;
  const float* src = cb + (size_t)c * DD;
  float s = 0.0f;
  #pragma unroll
  for (int i = 0; i < 4; ++i) {
    float v = src[t + i * 128];
    cb16[(size_t)c * DD + t + i * 128] = f2bf(v);
    s += v * v;
  }
  #pragma unroll
  for (int m = 16; m >= 1; m >>= 1) s += __shfl_xor(s, m, 32);
  if ((t & 31) == 0) red[t >> 5] = s;
  __syncthreads();
  if (t == 0) cnorm[c] = red[0] + red[1] + red[2] + red[3];
}

// transpose conv_w [kw, ci, co] f32 -> [kw, co, ci] bf16 (B columns contiguous)
__global__ __launch_bounds__(128)
void prep_convw_kernel(const float* __restrict__ w,
                       unsigned short* __restrict__ wT) {
  const int kw = blockIdx.x >> 9, co = blockIdx.x & 511, t = threadIdx.x;
  #pragma unroll
  for (int i = 0; i < 4; ++i) {
    int ci = t + i * 128;
    wT[(size_t)kw * DD * DD + (size_t)co * DD + ci] =
        f2bf(w[(size_t)kw * DD * DD + (size_t)ci * DD + co]);
  }
}

__global__ __launch_bounds__(256)
void cvt_qout_kernel(const float* __restrict__ q, unsigned short* __restrict__ q16) {
  size_t base = (size_t)blockIdx.x * 2048;
  #pragma unroll
  for (int i = 0; i < 8; ++i) {
    size_t e = base + threadIdx.x + (size_t)i * 256;
    q16[e] = f2bf(q[e]);
  }
}

// ===========================================================================
// One VQ stage: 16 rows per block, 4 waves x 256 codes each.
// dist = ||c||^2 - 2 r.c  (row norm dropped: constant under argmin)
__global__ __launch_bounds__(128)
void vq_stage_kernel(const float* __restrict__ cb_f32,        // [K,D] this stage
                     const unsigned short* __restrict__ cb16, // [K,D] bf16
                     const float* __restrict__ cnorm,         // [K]
                     float* __restrict__ residual,            // [B*T, D]
                     float* __restrict__ qout,                // [B*T, D]
                     float* __restrict__ idx_out,             // [B*T] (as float)
                     float* __restrict__ loss_out,
                     int first) {
  __shared__ unsigned short Ash[16][520];   // +8 pad: kill bank conflicts
  __shared__ float redv[4][16];
  __shared__ int   redi[4][16];
  __shared__ int   finalIdx[16];
  __shared__ float redf[4];

  const int t = threadIdx.x, lane = t & 31, wave = t >> 5;
  const int mbase = blockIdx.x * 16;

  // stage residual tile into LDS as bf16 (float4 coalesced reads)
  const float4* r4 = (const float4*)(residual + (size_t)mbase * DD);
  #pragma unroll
  for (int i = 0; i < 16; ++i) {
    int e = t + i * 128;                    // 0..2047 float4s
    int row = e >> 7, col = (e & 127) * 4;
    float4 v = r4[e];
    Ash[row][col + 0] = f2bf(v.x);
    Ash[row][col + 1] = f2bf(v.y);
    Ash[row][col + 2] = f2bf(v.z);
    Ash[row][col + 3] = f2bf(v.w);
  }
  __syncthreads();

  float minv[8]; int mini[8];
  #pragma unroll
  for (int r = 0; r < 8; ++r) { minv[r] = 3.4e38f; mini[r] = 0; }

  const int lrow = lane & 15, lhi = lane >> 4;
  const unsigned short* aptr = &Ash[lrow][lhi << 4];

  for (int nt = 0; nt < 16; ++nt) {
    const int nb = wave * 256 + nt * 16;
    v8f acc = {0.f, 0.f, 0.f, 0.f, 0.f, 0.f, 0.f, 0.f};
    // B column (one code row) is contiguous along K: single 32B run per lane
    const unsigned short* bptr = cb16 + (size_t)(nb + lrow) * DD + (lhi << 4);
    #pragma unroll
    for (int kk = 0; kk < DD; kk += 32) {
      BFrag a, b;
      a.h[0] = *(const v8bf*)(aptr + kk);        // K run 0..7 (or 16..23)
      a.h[1] = *(const v8bf*)(aptr + kk + 8);    // K run 8..15 (or 24..31)
      b.v    = *(const v16bf*)(bptr + kk);       // K 0..15 (or 16..31)
      acc = __builtin_amdgcn_wmma_f32_16x16x32_bf16(
          false, a.v, false, b.v, (short)0, acc, false, false);
    }
    const float cn = cnorm[nb + lrow];
    const int ci = nb + lrow;
    #pragma unroll
    for (int r = 0; r < 8; ++r) {
      float d = cn - 2.0f * acc[r];
      if (d < minv[r] || (d == minv[r] && ci < mini[r])) { minv[r] = d; mini[r] = ci; }
    }
  }

  // argmin across 16 column-lanes (xor<=8 stays within each 16-lane half)
  #pragma unroll
  for (int r = 0; r < 8; ++r) {
    #pragma unroll
    for (int m = 8; m >= 1; m >>= 1) {
      float ov = __shfl_xor(minv[r], m, 32);
      int   oi = __shfl_xor(mini[r], m, 32);
      if (ov < minv[r] || (ov == minv[r] && oi < mini[r])) { minv[r] = ov; mini[r] = oi; }
    }
  }
  if (lrow == 0) {
    #pragma unroll
    for (int r = 0; r < 8; ++r) {            // C layout: vgpr r -> row lhi*8+r
      redv[wave][lhi * 8 + r] = minv[r];
      redi[wave][lhi * 8 + r] = mini[r];
    }
  }
  __syncthreads();
  if (t < 16) {                               // cross-wave argmin, 16 rows
    float bv = redv[0][t]; int bi = redi[0][t];
    #pragma unroll
    for (int w = 1; w < 4; ++w) {
      float ov = redv[w][t]; int oi = redi[w][t];
      if (ov < bv || (ov == bv && oi < bi)) { bv = ov; bi = oi; }
    }
    finalIdx[t] = bi;
    idx_out[mbase + t] = (float)bi;
  }
  __syncthreads();

  // gather q (fp32), update residual/quantized_out, commit loss
  const int row = t >> 3, c0 = (t & 7) * 64;
  const int gi = finalIdx[row];
  const float4* qrow = (const float4*)(cb_f32 + (size_t)gi * DD + c0);
  float4* rrow = (float4*)(residual + (size_t)(mbase + row) * DD + c0);
  float4* orow = (float4*)(qout + (size_t)(mbase + row) * DD + c0);
  float lsum = 0.0f;
  #pragma unroll 4
  for (int i = 0; i < 16; ++i) {
    float4 q = qrow[i], rv = rrow[i];
    float4 nr = make_float4(rv.x - q.x, rv.y - q.y, rv.z - q.z, rv.w - q.w);
    rrow[i] = nr;
    if (first) orow[i] = q;
    else {
      float4 o = orow[i];
      orow[i] = make_float4(o.x + q.x, o.y + q.y, o.z + q.z, o.w + q.w);
    }
    lsum += nr.x * nr.x + nr.y * nr.y + nr.z * nr.z + nr.w * nr.w;
  }
  #pragma unroll
  for (int m = 16; m >= 1; m >>= 1) lsum += __shfl_xor(lsum, m, 32);
  if (lane == 0) redf[wave] = lsum;
  __syncthreads();
  if (t == 0)
    atomicAdd(loss_out, (redf[0] + redf[1] + redf[2] + redf[3]) *
                        (0.25f / (float)(BB * TT * DD)));
}

// ===========================================================================
// Conv1D stride 2, k=5, SAME (pad_lo=1) + exact GELU.  M-tile = 16 out rows;
// the 35 needed input rows staged once in LDS; 8 waves x 64 cols each.
__global__ __launch_bounds__(256)
void conv_gelu_kernel(const unsigned short* __restrict__ q16, // [B*T, D] bf16
                      const unsigned short* __restrict__ wT,  // [5, co, ci] bf16
                      float* __restrict__ y) {                // [B, TOUT, D]
  __shared__ unsigned short Ash[35][520];
  const int t = threadIdx.x, lane = t & 31, wave = t >> 5;
  const int b = blockIdx.x >> 7, tobase = (blockIdx.x & 127) * 16;
  const int inbase = 2 * tobase - 1;          // local row 0 -> this input pos

  #pragma unroll
  for (int i = 0; i < 18; ++i) {              // 35*128 ushort4s / 256 threads
    int e = t + i * 256;
    if (e < 35 * 128) {
      int row = e >> 7, col = (e & 127) * 4;
      int ip = inbase + row;
      ushort4 v = make_ushort4(0, 0, 0, 0);
      if (ip >= 0 && ip < TT)
        v = *(const ushort4*)(q16 + ((size_t)b * TT + ip) * DD + col);
      *(ushort4*)&Ash[row][col] = v;
    }
  }
  __syncthreads();

  v8f acc[4] = {{0.f,0.f,0.f,0.f,0.f,0.f,0.f,0.f}, {0.f,0.f,0.f,0.f,0.f,0.f,0.f,0.f},
                {0.f,0.f,0.f,0.f,0.f,0.f,0.f,0.f}, {0.f,0.f,0.f,0.f,0.f,0.f,0.f,0.f}};
  const int lrow = lane & 15, lhi = lane >> 4;
  const int ncol0 = wave * 64;

  #pragma unroll
  for (int kw = 0; kw < 5; ++kw) {
    const unsigned short* wbase = wT + (size_t)kw * DD * DD + (lhi << 4);
    const unsigned short* aptr = &Ash[2 * lrow + kw][lhi << 4];
    for (int kk = 0; kk < DD; kk += 32) {
      BFrag a;
      a.h[0] = *(const v8bf*)(aptr + kk);
      a.h[1] = *(const v8bf*)(aptr + kk + 8);
      #pragma unroll
      for (int nt = 0; nt < 4; ++nt) {        // A fragment reused 4x
        const int co = ncol0 + nt * 16 + lrow;
        BFrag bf;
        bf.v = *(const v16bf*)(wbase + (size_t)co * DD + kk);
        acc[nt] = __builtin_amdgcn_wmma_f32_16x16x32_bf16(
            false, a.v, false, bf.v, (short)0, acc[nt], false, false);
      }
    }
  }

  #pragma unroll
  for (int nt = 0; nt < 4; ++nt) {
    const int co = ncol0 + nt * 16 + lrow;
    #pragma unroll
    for (int r = 0; r < 8; ++r) {
      const int m = lhi * 8 + r;
      float x = acc[nt][r];
      float g = 0.5f * x * (1.0f + erff(x * 0.70710678118f));   // exact GELU
      y[((size_t)b * TOUT + tobase + m) * DD + co] = g;
    }
  }
}

// ===========================================================================
extern "C" void kernel_launch(void* const* d_in, const int* in_sizes, int n_in,
                              void* d_out, int out_size, void* d_ws, size_t ws_size,
                              hipStream_t stream) {
  const float* x  = (const float*)d_in[0];
  const float* cb = (const float*)d_in[1];
  const float* w  = (const float*)d_in[2];
  float* out = (float*)d_out;
  char*  ws  = (char*)d_ws;

  float*          residual = (float*)(ws + RES_OFF);
  unsigned short* cb16     = (unsigned short*)(ws + CB16_OFF);
  float*          cnorm    = (float*)(ws + CNORM_OFF);
  unsigned short* wT       = (unsigned short*)(ws + WT_OFF);
  unsigned short* q16      = (unsigned short*)(ws + Q16_OFF);

  float* y    = out + Y_OFF;
  float* qout = out + QOUT_OFF;
  float* idxo = out + IDX_OFF;
  float* loss = out + LOSS_OFF;

  hipMemcpyAsync(residual, x, (size_t)BB * TT * DD * sizeof(float),
                 hipMemcpyDeviceToDevice, stream);
  zero_loss_kernel<<<1, 1, 0, stream>>>(loss);
  prep_codebook_kernel<<<QQ * KK_, 128, 0, stream>>>(cb, cb16, cnorm);
  prep_convw_kernel<<<5 * DD, 128, 0, stream>>>(w, wT);

  for (int qi = 0; qi < QQ; ++qi) {
    vq_stage_kernel<<<(BB * TT) / 16, 128, 0, stream>>>(
        cb + (size_t)qi * KK_ * DD, cb16 + (size_t)qi * KK_ * DD,
        cnorm + qi * KK_, residual, qout, idxo + (size_t)qi * BB * TT, loss,
        qi == 0 ? 1 : 0);
  }

  cvt_qout_kernel<<<(BB * TT * DD) / 2048, 256, 0, stream>>>(qout, q16);
  conv_gelu_kernel<<<(BB * TOUT) / 16, 256, 0, stream>>>(q16, wT, y);
}
